// ColumnAttention_1116691497259
// MI455X (gfx1250) — matrix operand order repaired and checked by reading, not compiled, not run on hardware
//
#include <hip/hip_runtime.h>
#include <hip/hip_bf16.h>

// ColumnAttention for MI455X (gfx1250, wave32, WMMA bf16 16x16x32 + TDM).
//   x:        [16, 1024, 384] f32
//   q_weight: [1024, 64, 384] f32   q_bias: [1024, 64] f32
//   k_weight: [1024, 64, 384] f32   attn_bias: [1024, 1024] f32
// Outputs (concatenated in d_out): out [16,1024,384] f32, attn [16,1024,1024] f32.
// Workspace: Q,K bf16 (2 MB each) + x bf16 (12 MB) = 16 MB.

#define BATCH 16
#define SEQ   1024
#define DIM   384
#define QK    64

typedef __attribute__((ext_vector_type(16))) __bf16 v16bf;
typedef __attribute__((ext_vector_type(8)))  __bf16 v8bf;
typedef __attribute__((ext_vector_type(8)))  float  v8f;
typedef __attribute__((ext_vector_type(4)))  unsigned int v4u;
typedef __attribute__((ext_vector_type(8)))  int    v8i;
typedef __attribute__((ext_vector_type(4)))  int    v4i;

// A-fragment (16x32, 16-bit) element->K map per CDNA5 ISA 7.12.2.
__device__ __forceinline__ int kmapA(int e, int half) {
    return ((e & 8) << 1) + half * 8 + (e & 7);
}

// ---------------------------------------------------------------------------
// TDM: DMA a (32 x 384) bf16 tile (rows contiguous, row stride = DIM elems)
// from global memory into LDS at byte offset lds_off.  D# per ISA ch.8.
// ---------------------------------------------------------------------------
__device__ __forceinline__ void tdm_load_x_slice(const __bf16* gptr, unsigned lds_off) {
    const unsigned long long ga = (unsigned long long)(size_t)gptr;
    v4u g0;
    g0[0] = 1u;                                          // count=1, user D#
    g0[1] = lds_off;                                     // lds_addr (bytes)
    g0[2] = (unsigned)(ga & 0xFFFFFFFFu);                // global_addr[31:0]
    g0[3] = (unsigned)((ga >> 32) & 0x01FFFFFFu)         // global_addr[56:32]
          | 0x80000000u;                                 // type=2 ("image")
    v8i g1;
    g1[0] = 0x10000;                                     // data_size=1 (2 bytes)
    g1[1] = (DIM & 0xFFFF) << 16;                        // tensor_dim0 lo16
    g1[2] = (SEQ & 0xFFFF) << 16;                        // dim0 hi | tensor_dim1 lo16
    g1[3] = DIM << 16;                                   // dim1 hi | tile_dim0=384
    g1[4] = 32;                                          // tile_dim1=32, tile_dim2=0
    g1[5] = DIM;                                         // tensor_dim0_stride lo32
    g1[6] = ((DIM * SEQ) & 0xFFFF) << 16;                // stride0 hi | stride1 lo16
    g1[7] = (DIM * SEQ) >> 16;                           // stride1 [47:16]
    v4i gz = {0, 0, 0, 0};                               // 2-D: groups 2/3 unused
#if __clang_major__ >= 23
    v8i gz8 = {0, 0, 0, 0, 0, 0, 0, 0};
    __builtin_amdgcn_tensor_load_to_lds(g0, g1, gz, gz, gz8, 0);
#else
    __builtin_amdgcn_tensor_load_to_lds(g0, g1, gz, gz, 0);
#endif
}

// ---------------------------------------------------------------------------
// Kernel 0: one-time x f32 -> bf16 (feeds kernel-1 A-operand and kernel-2
// phase-3 B-operand; halves x bytes and removes inner-loop converts).
// ---------------------------------------------------------------------------
__global__ __launch_bounds__(256) void xcvt_kernel(
    const float* __restrict__ x, __bf16* __restrict__ xb)
{
    const size_t i = ((size_t)blockIdx.x * 256 + threadIdx.x) * 8;
    const float4 a = *(const float4*)(x + i);
    const float4 c = *(const float4*)(x + i + 4);
    v8bf o;
    o[0] = (__bf16)a.x; o[1] = (__bf16)a.y; o[2] = (__bf16)a.z; o[3] = (__bf16)a.w;
    o[4] = (__bf16)c.x; o[5] = (__bf16)c.y; o[6] = (__bf16)c.z; o[7] = (__bf16)c.w;
    *(v8bf*)(xb + i) = o;
}

// ---------------------------------------------------------------------------
// Kernel 1: per-position untied projections (M=16=BATCH, N=64, K=384 per n).
// Wave w computes qk-dim tile [16w,16w+16) for BOTH Q and K; 12 K-steps.
// Bandwidth-bound on the 200 MB of weights -> stream f32 weights once.
// ---------------------------------------------------------------------------
__global__ __launch_bounds__(128) void qk_proj_kernel(
    const __bf16* __restrict__ x_bf, const float* __restrict__ qw,
    const float* __restrict__ qb, const float* __restrict__ kw,
    __bf16* __restrict__ q_ws, __bf16* __restrict__ k_ws)
{
    const int n    = blockIdx.x;
    const int tid  = threadIdx.x;
    const int wave = tid >> 5;
    const int lane = tid & 31;
    const int half = lane >> 4;
    const int l16  = lane & 15;

    const int dBase = wave * 16;

    v8f accq = {};
    v8f acck = {};

    const __bf16* xrow  = x_bf + ((size_t)l16 * SEQ + n) * DIM;      // A row = batch
    const float*  qwrow = qw + ((size_t)n * QK + dBase + l16) * DIM; // B col = qk dim
    const float*  kwrow = kw + ((size_t)n * QK + dBase + l16) * DIM;

    for (int c0 = 0; c0 < DIM; c0 += 32) {
        v16bf a, bq, bk;
#pragma unroll
        for (int e = 0; e < 16; ++e)
            a[e] = xrow[c0 + half * 16 + e];
        const float* qp = qwrow + c0 + half * 16;   // 16 contiguous floats
        const float* kp = kwrow + c0 + half * 16;
        __builtin_prefetch(qp + 32, 0, 1);          // global_prefetch next K-step
        __builtin_prefetch(kp + 32, 0, 1);
#pragma unroll
        for (int e = 0; e < 16; ++e) {
            bq[e] = (__bf16)qp[e];
            bk[e] = (__bf16)kp[e];
        }
        accq = __builtin_amdgcn_wmma_f32_16x16x32_bf16(false, a, false, bq,
                                                       (short)0, accq, false, false);
        acck = __builtin_amdgcn_wmma_f32_16x16x32_bf16(false, a, false, bk,
                                                       (short)0, acck, false, false);
    }

    const float scale = 0.05103103630798287f;       // DIM^-0.5
    const int   d     = dBase + l16;
    const float bias  = qb[n * QK + d];
#pragma unroll
    for (int r = 0; r < 8; ++r) {
        const int    b   = half * 8 + r;            // C/D layout: m = half*8 + r
        const size_t idx = ((size_t)b * SEQ + n) * QK + d;
        q_ws[idx] = (__bf16)(scale * (accq[r] + bias));
        k_ws[idx] = (__bf16)(acck[r]);
    }
}

// ---------------------------------------------------------------------------
// Kernel 2: one block per (b, 16-row n-tile); 4 waves.
//   Phase 1: S = Q_tile @ K^T + bias -> 64 KB LDS
//   Phase 2: row softmax (8 lanes/row, in-wave shuffles) -> attn out + LDS
//   Phase 3: out_tile = P @ x_bf[b]; x K-slices (32x384 bf16, 24 KB) staged
//            into LDS by the Tensor Data Mover, double-buffered so the DMA of
//            slice i+1 overlaps the 6 WMMAs/wave of slice i.
// ---------------------------------------------------------------------------
__global__ __launch_bounds__(128) void attn_kernel(
    const __bf16* __restrict__ x_bf, const __bf16* __restrict__ q_ws,
    const __bf16* __restrict__ k_ws, const float* __restrict__ attn_bias,
    float* __restrict__ out, float* __restrict__ attn_out)
{
    __shared__ __align__(16) float  sc[16 * SEQ];      // 64 KB scores -> probs
    __shared__ __align__(16) __bf16 xs[2][32 * DIM];   // 2 x 24 KB x K-slices

    const int b    = blockIdx.x >> 6;
    const int nt   = blockIdx.x & 63;
    const int tid  = threadIdx.x;
    const int wave = tid >> 5;
    const int lane = tid & 31;
    const int half = lane >> 4;
    const int l16  = lane & 15;

    // ---- Phase 1: Q@K^T + bias --------------------------------------------
    v16bf aq0, aq1;
    {
        const __bf16* qrow = q_ws + ((size_t)b * SEQ + nt * 16 + l16) * QK;
#pragma unroll
        for (int e = 0; e < 16; ++e) {
            const int k = half * 16 + e;
            aq0[e] = qrow[k];
            aq1[e] = qrow[32 + k];
        }
    }
    for (int mt = wave * 16; mt < wave * 16 + 16; ++mt) {
        const __bf16* krow = k_ws + ((size_t)b * SEQ + mt * 16 + l16) * QK;
        const int kb = half * 16;
        v16bf bk0, bk1;
#pragma unroll
        for (int e = 0; e < 16; ++e) {
            bk0[e] = krow[kb + e];
            bk1[e] = krow[32 + kb + e];
        }
        v8f cs = {};
        cs = __builtin_amdgcn_wmma_f32_16x16x32_bf16(false, aq0, false, bk0,
                                                     (short)0, cs, false, false);
        cs = __builtin_amdgcn_wmma_f32_16x16x32_bf16(false, aq1, false, bk1,
                                                     (short)0, cs, false, false);
#pragma unroll
        for (int r = 0; r < 8; ++r) {
            const int row = half * 8 + r;
            const int n   = nt * 16 + row;
            const int m   = mt * 16 + l16;
            sc[row * SEQ + m] = cs[r] + attn_bias[(size_t)n * SEQ + m];
        }
    }
    __syncthreads();

    // ---- Phase 2: softmax per row -----------------------------------------
    {
        const int row = tid >> 3;
        const int g   = tid & 7;                   // 8 lanes/row, same wave
        float* srow = sc + row * SEQ;
        const int c0 = g * 128;

        float mx = -3.4e38f;
        for (int c = c0; c < c0 + 128; ++c) mx = fmaxf(mx, srow[c]);
#pragma unroll
        for (int off = 4; off >= 1; off >>= 1)
            mx = fmaxf(mx, __shfl_xor(mx, off, 8));

        float sum = 0.0f;
        for (int c = c0; c < c0 + 128; ++c) {
            const float e = __expf(srow[c] - mx);
            srow[c] = e;
            sum += e;
        }
#pragma unroll
        for (int off = 4; off >= 1; off >>= 1)
            sum += __shfl_xor(sum, off, 8);
        const float inv = 1.0f / sum;

        const int n = nt * 16 + row;
        float* arow = attn_out + ((size_t)b * SEQ + n) * SEQ;
        for (int c = c0; c < c0 + 128; ++c) {
            const float p = srow[c] * inv;
            srow[c] = p;
            arow[c] = p;
        }
    }
    __syncthreads();

    // ---- Phase 3: out_tile = P @ x_bf[b], TDM double-buffered -------------
    v8f acc[6];
#pragma unroll
    for (int t = 0; t < 6; ++t) acc[t] = (v8f){};

    const __bf16*  xbf_b   = x_bf + (size_t)b * SEQ * DIM;
    const unsigned xs_base = (unsigned)(size_t)&xs[0][0];  // LDS byte offset
    const unsigned xs_step = 32 * DIM * 2;                 // 24 KB per buffer

    if (wave == 0) {                              // prologue: stage slice 0
        tdm_load_x_slice(xbf_b, xs_base);
        __builtin_amdgcn_s_wait_tensorcnt(0);
    }
    __syncthreads();

    for (int it = 0; it < SEQ / 32; ++it) {
        const int cur = it & 1;
        if (wave == 0 && it + 1 < SEQ / 32)       // DMA next slice (overlaps WMMA)
            tdm_load_x_slice(xbf_b + (size_t)(it + 1) * 32 * DIM,
                             xs_base + (cur ^ 1) * xs_step);

        v16bf ap;
#pragma unroll
        for (int e = 0; e < 16; ++e)
            ap[e] = (__bf16)sc[l16 * SEQ + it * 32 + half * 16 + e];

        const __bf16* xsb = xs[cur];
#pragma unroll
        for (int t = 0; t < 6; ++t) {
            const int c = (wave * 6 + t) * 16 + l16;
            v16bf bx;
#pragma unroll
            for (int e = 0; e < 16; ++e)
                bx[e] = xsb[(half * 16 + e) * DIM + c];
            acc[t] = __builtin_amdgcn_wmma_f32_16x16x32_bf16(false, ap, false, bx,
                                                             (short)0, acc[t],
                                                             false, false);
        }

        if (wave == 0) __builtin_amdgcn_s_wait_tensorcnt(0);
        __syncthreads();                          // publish next buffer
    }

#pragma unroll
    for (int t = 0; t < 6; ++t) {
        const int c = (wave * 6 + t) * 16 + l16;
#pragma unroll
        for (int r = 0; r < 8; ++r) {
            const int n = nt * 16 + half * 8 + r;
            out[((size_t)b * SEQ + n) * DIM + c] = acc[t][r];
        }
    }
}

extern "C" void kernel_launch(void* const* d_in, const int* in_sizes, int n_in,
                              void* d_out, int out_size, void* d_ws, size_t ws_size,
                              hipStream_t stream) {
    const float* x  = (const float*)d_in[0];
    const float* qw = (const float*)d_in[1];
    const float* qb = (const float*)d_in[2];
    const float* kw = (const float*)d_in[3];
    const float* ab = (const float*)d_in[4];

    float* out  = (float*)d_out;                                   // [16,1024,384]
    float* attn = out + (size_t)BATCH * SEQ * DIM;                 // [16,1024,1024]

    __bf16* q_ws = (__bf16*)d_ws;                                  // 2 MB
    __bf16* k_ws = q_ws + (size_t)BATCH * SEQ * QK;                // 2 MB
    __bf16* x_bf = k_ws + (size_t)BATCH * SEQ * QK;                // 12 MB

    const size_t xelems = (size_t)BATCH * SEQ * DIM;
    xcvt_kernel<<<(int)(xelems / (256 * 8)), 256, 0, stream>>>(x, x_bf);
    qk_proj_kernel<<<SEQ, 128, 0, stream>>>(x_bf, qw, qb, kw, q_ws, k_ws);
    attn_kernel<<<BATCH * (SEQ / 16), 128, 0, stream>>>(x_bf, q_ws, k_ws, ab, out, attn);
}
